// VQVAE_43026982371811
// MI455X (gfx1250) — matrix operand order repaired and checked by reading, not compiled, not run on hardware
//
#include <hip/hip_runtime.h>
#include <hip/hip_bf16.h>

typedef _Float16 v16h __attribute__((ext_vector_type(16)));
typedef _Float16 v8h  __attribute__((ext_vector_type(8)));
typedef float    v8f  __attribute__((ext_vector_type(8)));

#define BN_EPS 1e-5f

// ---------------------------------------------------------------- zero stats
__global__ void k_zero(float* __restrict__ p, int n) {
  int i = blockIdx.x * 256 + threadIdx.x;
  if (i < n) p[i] = 0.f;
}

// ------------------------------------- codebook prep: f16 copy (K padded 256)
// and cnorm[k] = 0.5 * ||codex[k]||^2
__global__ void k_codex_prep(const float* __restrict__ codex,
                             float* __restrict__ cnorm,
                             _Float16* __restrict__ ch) {
  int k = blockIdx.x;      // 1024 entries
  int j = threadIdx.x;     // 256 (padded K)
  float v = (j < 240) ? codex[k * 240 + j] : 0.f;
  ch[k * 256 + j] = (_Float16)v;
  __shared__ float red[256];
  red[j] = 0.5f * v * v;
  __syncthreads();
  for (int s = 128; s > 0; s >>= 1) {
    if (j < s) red[j] += red[j + s];
    __syncthreads();
  }
  if (j == 0) cnorm[k] = red[0];
}

// ------------------------------- conv1d k=3 stride=2 pad=1 + bias + BN stats
// grid: (ceil(Lout/256), Cout, N); block 256
__global__ void k_conv_s2(const float* __restrict__ x, const float* __restrict__ w,
                          const float* __restrict__ bias, float* __restrict__ y,
                          float* __restrict__ stats,
                          int Cin, int Lin, int Cout, int Lout) {
  int c = blockIdx.y, n = blockIdx.z, tid = threadIdx.x;
  __shared__ float wl[720];
  for (int i = tid; i < Cin * 3; i += 256) wl[i] = w[c * Cin * 3 + i];  // (O,I,K) slice
  __syncthreads();
  int l = blockIdx.x * 256 + tid;
  float acc = 0.f;
  if (l < Lout) {
    acc = bias[c];
    int p0 = 2 * l - 1;
    const float* xn = x + (long)n * Cin * Lin;
    for (int ci = 0; ci < Cin; ++ci) {
      const float* xc = xn + (long)ci * Lin;
      const float* wc = wl + ci * 3;
#pragma unroll
      for (int k = 0; k < 3; ++k) {
        int p = p0 + k;
        if (p >= 0 && p < Lin) acc += xc[p] * wc[k];
      }
    }
    y[((long)n * Cout + c) * Lout + l] = acc;
  }
  __shared__ float rs[256], rq[256];
  float v = (l < Lout) ? acc : 0.f;
  rs[tid] = v; rq[tid] = v * v;
  __syncthreads();
  for (int s = 128; s > 0; s >>= 1) {
    if (tid < s) { rs[tid] += rs[tid + s]; rq[tid] += rq[tid + s]; }
    __syncthreads();
  }
  if (tid == 0) {
    atomicAdd(&stats[2 * c], rs[0]);
    atomicAdd(&stats[2 * c + 1], rq[0]);
  }
}

// --------- convT1d k=3 s=2 pad=1 outpad=1 (+bias, optional BN stats)
// Lout = 2*Lin.  grid: (ceil(Lout/256), Cout, N)
__global__ void k_convT(const float* __restrict__ x, const float* __restrict__ w,
                        const float* __restrict__ bias, float* __restrict__ y,
                        float* __restrict__ stats,
                        int Cin, int Lin, int Cout, int with_stats) {
  int Lout = 2 * Lin;
  int c = blockIdx.y, n = blockIdx.z, tid = threadIdx.x;
  __shared__ float wl[720];
  for (int i = tid; i < Cin * 3; i += 256) {
    int ci = i / 3, k = i % 3;                     // w layout (I,O,K)
    wl[i] = w[((long)ci * Cout + c) * 3 + k];
  }
  __syncthreads();
  int t = blockIdx.x * 256 + tid;
  float acc = 0.f;
  if (t < Lout) {
    acc = bias[c];
    const float* xn = x + (long)n * Cin * Lin;
    if ((t & 1) == 0) {
      int p = t >> 1;
      for (int ci = 0; ci < Cin; ++ci) acc += xn[(long)ci * Lin + p] * wl[ci * 3 + 1];
    } else {
      int p0 = (t - 1) >> 1, p1 = (t + 1) >> 1;
      if (p1 < Lin) {
        for (int ci = 0; ci < Cin; ++ci) {
          const float* xc = xn + (long)ci * Lin;
          acc += xc[p0] * wl[ci * 3 + 2] + xc[p1] * wl[ci * 3 + 0];
        }
      } else {
        for (int ci = 0; ci < Cin; ++ci) acc += xn[(long)ci * Lin + p0] * wl[ci * 3 + 2];
      }
    }
    y[((long)n * Cout + c) * Lout + t] = acc;
  }
  if (with_stats) {
    __shared__ float rs[256], rq[256];
    float v = (t < Lout) ? acc : 0.f;
    rs[tid] = v; rq[tid] = v * v;
    __syncthreads();
    for (int s = 128; s > 0; s >>= 1) {
      if (tid < s) { rs[tid] += rs[tid + s]; rq[tid] += rq[tid + s]; }
      __syncthreads();
    }
    if (tid == 0) {
      atomicAdd(&stats[2 * c], rs[0]);
      atomicAdd(&stats[2 * c + 1], rq[0]);
    }
  }
}

// -------------------------------------------------- BN (batch stats) + ReLU
__global__ void k_bn_relu(const float* __restrict__ y, const float* __restrict__ stats,
                          const float* __restrict__ g, const float* __restrict__ be,
                          float* __restrict__ out, int C, int L, long total,
                          float invNL, int do_relu) {
  long i = (long)blockIdx.x * 256 + threadIdx.x;
  if (i >= total) return;
  int c = (int)((i / L) % C);
  float mu  = stats[2 * c] * invNL;
  float var = stats[2 * c + 1] * invNL - mu * mu;
  float sc  = g[c] * rsqrtf(var + BN_EPS);
  float v   = (y[i] - mu) * sc + be[c];
  out[i] = (do_relu && v < 0.f) ? 0.f : v;
}

// ------------------------------------------------ VQ nearest-neighbor + gather
// score(m,n) = 0.5*||c_n||^2 - f_m . c_n   (argmin equals argmin of L2 dist)
// GEMM via v_wmma_f32_16x16x32_f16.
// Block = 32 rows of flat (two 16-row A tiles -> each B fragment feeds 2 WMMAs,
// halving L2 codebook traffic vs 16-row tiles); 8 waves sweep N=1024.
#define ASTR 264   // f16 LDS row stride: conflict-free for 16-byte ds loads
__global__ void __launch_bounds__(256)
k_vq(const float* __restrict__ flat, const _Float16* __restrict__ ch,
     const float* __restrict__ codex, const float* __restrict__ cnorm,
     float* __restrict__ quant) {
  __shared__ _Float16 As[32 * ASTR];
  __shared__ float candS[32 * 16 * 8];
  __shared__ int   candI[32 * 16 * 8];
  __shared__ int   sIdx[32];
  int tid = threadIdx.x;
  int m0 = blockIdx.x * 32;

  // stage A tile (32 x 256 f16, zero-padded K)
  for (int i = tid; i < 32 * 256; i += 256) {
    int r = i >> 8, cc = i & 255;
    As[r * ASTR + cc] = (cc < 240) ? (_Float16)flat[(long)(m0 + r) * 240 + cc]
                                   : (_Float16)0.f;
  }
  __syncthreads();

  int wave = tid >> 5, lane = tid & 31;
  int half = lane >> 4, lo = lane & 15;
  const _Float16* arow0 = As + lo * ASTR + half * 16;          // rows 0..15
  const _Float16* arow1 = As + (16 + lo) * ASTR + half * 16;   // rows 16..31

  float best0[8], best1[8]; int bidx0[8], bidx1[8];
#pragma unroll
  for (int q = 0; q < 8; ++q) {
    best0[q] = 3.0e38f; bidx0[q] = 0;
    best1[q] = 3.0e38f; bidx1[q] = 0;
  }

  for (int nt = wave; nt < 64; nt += 8) {           // 64 N-tiles of 16
    int nidx = nt * 16 + lo;
    const _Float16* brow = ch + (long)nidx * 256 + half * 16;
    v8f acc0 = {}, acc1 = {};
#pragma unroll
    for (int kk = 0; kk < 256; kk += 32) {
      v16h b = *(const v16h*)(brow + kk);           // contiguous 16 K-values (shared)
      v8h a00 = *(const v8h*)(arow0 + kk);
      v8h a01 = *(const v8h*)(arow0 + kk + 8);
      v16h a0 = __builtin_shufflevector(a00, a01, 0,1,2,3,4,5,6,7,8,9,10,11,12,13,14,15);
      acc0 = __builtin_amdgcn_wmma_f32_16x16x32_f16(
          false, a0, false, b, (short)0, acc0, false, false);
      v8h a10 = *(const v8h*)(arow1 + kk);
      v8h a11 = *(const v8h*)(arow1 + kk + 8);
      v16h a1 = __builtin_shufflevector(a10, a11, 0,1,2,3,4,5,6,7,8,9,10,11,12,13,14,15);
      acc1 = __builtin_amdgcn_wmma_f32_16x16x32_f16(
          false, a1, false, b, (short)0, acc1, false, false);
    }
    float cn = cnorm[nidx];
#pragma unroll
    for (int q = 0; q < 8; ++q) {                   // D row m = q + half*8, col n = lo
      float s0 = cn - acc0[q];
      if (s0 < best0[q]) { best0[q] = s0; bidx0[q] = nidx; }
      float s1 = cn - acc1[q];
      if (s1 < best1[q]) { best1[q] = s1; bidx1[q] = nidx; }
    }
  }
#pragma unroll
  for (int q = 0; q < 8; ++q) {
    int row0 = q + half * 8;
    int slot0 = (row0 * 16 + lo) * 8 + wave;
    candS[slot0] = best0[q]; candI[slot0] = bidx0[q];
    int row1 = 16 + q + half * 8;
    int slot1 = (row1 * 16 + lo) * 8 + wave;
    candS[slot1] = best1[q]; candI[slot1] = bidx1[q];
  }
  __syncthreads();
  if (tid < 32) {                                   // reduce 128 candidates per row
    float b = 3.0e38f; int bi = 0;
    for (int j = 0; j < 128; ++j) {
      float s = candS[tid * 128 + j];
      if (s < b) { b = s; bi = candI[tid * 128 + j]; }
    }
    sIdx[tid] = bi;
  }
  __syncthreads();
  // gather exact f32 codewords
  for (int i = tid; i < 32 * 240; i += 256) {
    int r = i / 240, cc = i % 240;
    quant[(long)(m0 + r) * 240 + cc] = codex[(long)sIdx[r] * 240 + cc];
  }
}

// ---------------------------------------------------------------------------
extern "C" void kernel_launch(void* const* d_in, const int* in_sizes, int n_in,
                              void* d_out, int out_size, void* d_ws, size_t ws_size,
                              hipStream_t stream) {
  (void)in_sizes; (void)n_in; (void)out_size; (void)ws_size;
  const float* x    = (const float*)d_in[0];
  const float* w1   = (const float*)d_in[1];
  const float* b1   = (const float*)d_in[2];
  const float* g1   = (const float*)d_in[3];
  const float* be1  = (const float*)d_in[4];
  const float* w2   = (const float*)d_in[5];
  const float* b2   = (const float*)d_in[6];
  const float* g2   = (const float*)d_in[7];
  const float* be2  = (const float*)d_in[8];
  const float* w3   = (const float*)d_in[9];
  const float* b3   = (const float*)d_in[10];
  const float* g3   = (const float*)d_in[11];
  const float* be3  = (const float*)d_in[12];
  const float* codex= (const float*)d_in[13];
  const float* dw1  = (const float*)d_in[14];
  const float* db1  = (const float*)d_in[15];
  const float* dg1  = (const float*)d_in[16];
  const float* dbe1 = (const float*)d_in[17];
  const float* dw2  = (const float*)d_in[18];
  const float* db2  = (const float*)d_in[19];
  const float* dg2  = (const float*)d_in[20];
  const float* dbe2 = (const float*)d_in[21];
  const float* dw3  = (const float*)d_in[22];
  const float* db3  = (const float*)d_in[23];

  float* ws = (float*)d_ws;
  float* A     = ws;                      // 7,864,320 floats
  float* Bb    = ws + 7864320L;           // 7,864,320 floats
  float* stats = ws + 15728640L;          // 1024 floats
  float* cnorm = ws + 15729664L;          // 1024 floats
  _Float16* ch = (_Float16*)(ws + 15730688L);  // 1024*256 halves

  float* out   = (float*)d_out;
  float* recon = out;                     // (128,14,1920)
  float* enc   = out + 3440640L;          // (128,240,240)
  float* quant = out + 10813440L;         // (128,240,240)

  k_zero<<<4, 256, 0, stream>>>(stats, 928);
  k_codex_prep<<<1024, 256, 0, stream>>>(codex, cnorm, ch);

  // ----- encoder -----
  k_conv_s2<<<dim3(4, 32, 128), 256, 0, stream>>>(x, w1, b1, A, stats + 0, 14, 1920, 32, 960);
  k_bn_relu<<<(int)((128L*32*960 + 255) / 256), 256, 0, stream>>>(
      A, stats + 0, g1, be1, Bb, 32, 960, 128L*32*960, 1.f / (128.f*960.f), 1);

  k_conv_s2<<<dim3(2, 64, 128), 256, 0, stream>>>(Bb, w2, b2, A, stats + 64, 32, 960, 64, 480);
  k_bn_relu<<<(int)((128L*64*480 + 255) / 256), 256, 0, stream>>>(
      A, stats + 64, g2, be2, Bb, 64, 480, 128L*64*480, 1.f / (128.f*480.f), 1);

  k_conv_s2<<<dim3(1, 240, 128), 256, 0, stream>>>(Bb, w3, b3, A, stats + 192, 64, 480, 240, 240);
  k_bn_relu<<<(int)((128L*240*240 + 255) / 256), 256, 0, stream>>>(
      A, stats + 192, g3, be3, enc, 240, 240, 128L*240*240, 1.f / (128.f*240.f), 1);

  // ----- VQ (WMMA GEMM + argmin + gather), 32-row M tiles -----
  k_vq<<<960, 256, 0, stream>>>(enc, ch, codex, cnorm, quant);

  // ----- decoder -----
  k_convT<<<dim3(2, 64, 128), 256, 0, stream>>>(quant, dw1, db1, A, stats + 672, 240, 240, 64, 1);
  k_bn_relu<<<(int)((128L*64*480 + 255) / 256), 256, 0, stream>>>(
      A, stats + 672, dg1, dbe1, Bb, 64, 480, 128L*64*480, 1.f / (128.f*480.f), 1);

  k_convT<<<dim3(4, 64, 128), 256, 0, stream>>>(Bb, dw2, db2, A, stats + 800, 64, 480, 64, 1);
  k_bn_relu<<<(int)((128L*64*960 + 255) / 256), 256, 0, stream>>>(
      A, stats + 800, dg2, dbe2, Bb, 64, 960, 128L*64*960, 1.f / (128.f*960.f), 1);

  k_convT<<<dim3(8, 14, 128), 256, 0, stream>>>(Bb, dw3, db3, recon, nullptr, 64, 960, 14, 0);
}
